// GINMLP_node_30305289241328
// MI455X (gfx1250) — matrix operand order repaired and checked by reading, not compiled, not run on hardware
//
#include <hip/hip_runtime.h>

#define NN 50000
#define EE 800000
#define DD 64
#define LL 4
#define BN_EPS 1e-5f

typedef __attribute__((ext_vector_type(16))) _Float16 v16h;
typedef __attribute__((ext_vector_type(8)))  float    v8f;

__device__ inline unsigned pack2h(float a, float b) {
    _Float16 ha = (_Float16)a, hb = (_Float16)b;
    unsigned short ua = __builtin_bit_cast(unsigned short, ha);
    unsigned short ub = __builtin_bit_cast(unsigned short, hb);
    return (unsigned)ua | ((unsigned)ub << 16);
}

__device__ inline v16h loadB(const unsigned* p) {
    union { unsigned u[8]; v16h v; } t;
#pragma unroll
    for (int i = 0; i < 8; ++i) t.u[i] = p[i];
    return t.v;
}

// ---------------- utility kernels ----------------
__global__ void zero_k(float* p, unsigned n) {
    unsigned i = blockIdx.x * blockDim.x + threadIdx.x;
    if (i < n) p[i] = 0.0f;
}
__global__ void copy_k(float* dst, const float* src, unsigned n) {
    unsigned i = blockIdx.x * blockDim.x + threadIdx.x;
    if (i < n) dst[i] = src[i];
}
__global__ void deg_k(const int* dst, float* deg) {
    unsigned i = blockIdx.x * blockDim.x + threadIdx.x;
    if (i < EE) atomicAdd(&deg[(unsigned)dst[i]], 1.0f);
}
__global__ void degfin_k(float* d) {
    unsigned i = blockIdx.x * blockDim.x + threadIdx.x;
    if (i < NN) d[i] = 1.0f / fmaxf(d[i], 1.0f);
}
__global__ void eastats_k(const float* a, float* stats) {
    unsigned i0 = blockIdx.x * blockDim.x + threadIdx.x;
    float s = 0.0f, q = 0.0f;
    for (unsigned i = i0; i < EE; i += gridDim.x * blockDim.x) {
        float v = a[i]; s += v; q += v * v;
    }
#pragma unroll
    for (int o = 16; o > 0; o >>= 1) { s += __shfl_xor(s, o, 32); q += __shfl_xor(q, o, 32); }
    if ((threadIdx.x & 31) == 0) { atomicAdd(&stats[0], s); atomicAdd(&stats[1], q); }
}

// ---------------- per-layer prep: collapse edge-encoder+BN, pack B matrices ----------------
__global__ void prep_k(int l, const float* W_enc, const float* W_r1, const float* g_r,
                       const float* W_r2, const float* W_m1, const float* W_m2,
                       const float* stats, float* mav, float* svec,
                       unsigned* Br2, unsigned* Bm1, unsigned* Bm2) {
    int tid = threadIdx.x; // 128 threads
    __shared__ float sva;
    if (tid == 0) {
        float ma = stats[0] * (1.0f / EE);
        float va = stats[1] * (1.0f / EE) - ma * ma;
        mav[0] = ma; mav[1] = va; sva = va;
    }
    __syncthreads();
    if (tid < 64) {
        const float* we  = W_enc + l * 64;
        const float* wr1 = W_r1 + l * 64 * 64;
        float u = 0.0f;
        for (int j = 0; j < 64; ++j) u += we[j] * wr1[j * 64 + tid];
        svec[tid] = u * g_r[l * 64 + tid] * rsqrtf(sva * u * u + BN_EPS);
    }
    // B layout per mat: K = 8*(lane/16) + (v>=4?16:0) + 2*(v&3) (+1) ; n = nt*16 + lane%16
    const float* wr2 = W_r2 + l * 64 * 64;
    for (int i = tid; i < 8 * 256; i += 128) {
        int mat = i >> 8, rem = i & 255, lane = rem >> 3, v = rem & 7;
        int nt = mat >> 1, kc = mat & 1;
        int n = nt * 16 + (lane & 15);
        int k = kc * 32 + ((lane >> 4) << 3) + ((v >= 4) ? 16 : 0) + 2 * (v & 3);
        Br2[i] = pack2h(wr2[k * 64 + n], wr2[(k + 1) * 64 + n]);
    }
    const float* wm1 = W_m1 + l * 64 * 128;
    for (int i = tid; i < 16 * 256; i += 128) {
        int mat = i >> 8, rem = i & 255, lane = rem >> 3, v = rem & 7;
        int nt = mat >> 1, kc = mat & 1;
        int n = nt * 16 + (lane & 15);
        int k = kc * 32 + ((lane >> 4) << 3) + ((v >= 4) ? 16 : 0) + 2 * (v & 3);
        Bm1[i] = pack2h(wm1[k * 128 + n], wm1[(k + 1) * 128 + n]);
    }
    const float* wm2 = W_m2 + l * 128 * 64;
    for (int i = tid; i < 16 * 256; i += 128) {
        int mat = i >> 8, rem = i & 255, lane = rem >> 3, v = rem & 7;
        int nt = mat >> 2, kc = mat & 3;
        int n = nt * 16 + (lane & 15);
        int k = kc * 32 + ((lane >> 4) << 3) + ((v >= 4) ? 16 : 0) + 2 * (v & 3);
        Bm2[i] = pack2h(wm2[k * 64 + n], wm2[(k + 1) * 64 + n]);
    }
}

// ---------------- edge kernel: r(a_e) -> WMMA x W_r2 -> relu(h[src]+rel) -> atomic scatter ----------------
__global__ __launch_bounds__(256) void edge_k(const float* __restrict__ ea, const int* __restrict__ ei,
                                              const float* __restrict__ mav, const float* __restrict__ svec,
                                              const float* __restrict__ bev, const float* __restrict__ br2b,
                                              const unsigned* __restrict__ Br2g,
                                              const float* __restrict__ h, float* __restrict__ agg) {
    __shared__ float ss[64], sbe[64], sb2[64];
    __shared__ unsigned sB[8 * 32 * 8];
    int tid = threadIdx.x;
    if (tid < 64) { ss[tid] = svec[tid]; sbe[tid] = bev[tid]; sb2[tid] = br2b[tid]; }
    for (int i = tid; i < 2048; i += 256) sB[i] = Br2g[i];
    __syncthreads();
    int wave = tid >> 5, L = tid & 31, row = L & 15, half = L >> 4;
    unsigned tile = blockIdx.x * 8u + wave;       // E/16 = 50000 tiles exactly
    unsigned e = tile * 16u + row;
    float am = ea[e] - mav[0];
    v16h A[2];
#pragma unroll
    for (int c = 0; c < 2; ++c)
#pragma unroll
        for (int v = 0; v < 8; ++v) {
            int kb = half * 8 + c * 32 + ((v >= 4) ? 16 : 0) + 2 * (v & 3);
            A[c][2 * v]     = (_Float16)fmaxf(fmaf(am, ss[kb],     sbe[kb]),     0.0f);
            A[c][2 * v + 1] = (_Float16)fmaxf(fmaf(am, ss[kb + 1], sbe[kb + 1]), 0.0f);
        }
    // rows handled by this lane in the C layout are contiguous edges: tile*16 + half*8 + v
    unsigned ebase = tile * 16u + half * 8u;
    unsigned soff[8], doff[8];
#pragma unroll
    for (int v = 0; v < 8; ++v) {
        soff[v] = (unsigned)ei[ebase + v] * 64u;
        doff[v] = (unsigned)ei[EE + ebase + v] * 64u;
    }
#pragma unroll
    for (int nt = 0; nt < 4; ++nt) {
        v8f acc = {};
        v16h B0 = loadB(sB + (((nt * 2 + 0) * 32 + L) * 8));
        v16h B1 = loadB(sB + (((nt * 2 + 1) * 32 + L) * 8));
        acc = __builtin_amdgcn_wmma_f32_16x16x32_f16(false, A[0], false, B0, (short)0, acc, false, false);
        acc = __builtin_amdgcn_wmma_f32_16x16x32_f16(false, A[1], false, B1, (short)0, acc, false, false);
        unsigned col = nt * 16u + row;
        float bb = sb2[col];
#pragma unroll
        for (int v = 0; v < 8; ++v) {
            float rel = acc[v] + bb;
            float msg = fmaxf(h[soff[v] + col] + rel, 0.0f);
            atomicAdd(&agg[doff[v] + col], msg);
        }
    }
}

// ---------------- node pass 1: z0 = (1+eps)h + agg/deg ; y = z0@W_m1+b ; BN column stats ----------------
__global__ __launch_bounds__(256) void node1_k(const float* __restrict__ h, const float* __restrict__ agg,
                                               const float* __restrict__ invd, const float* __restrict__ epsp,
                                               const unsigned* __restrict__ Bm1g, const float* __restrict__ bias,
                                               float* __restrict__ y, float* colsum, float* colsq) {
    __shared__ unsigned sB[16 * 32 * 8]; // 16KB
    __shared__ float sbias[128];
    int tid = threadIdx.x;
    for (int i = tid; i < 4096; i += 256) sB[i] = Bm1g[i];
    if (tid < 128) sbias[tid] = bias[tid];
    __syncthreads();
    int wave = tid >> 5, L = tid & 31, row = L & 15, half = L >> 4;
    unsigned tile = blockIdx.x * 8u + wave;
    if (tile >= NN / 16) return;
    unsigned nA = tile * 16u + row;
    float epsl = 1.0f + *epsp;
    float iv = invd[nA];
    const float* hp = h   + nA * 64u;
    const float* ap = agg + nA * 64u;
    v16h A[2];
#pragma unroll
    for (int c = 0; c < 2; ++c)
#pragma unroll
        for (int v = 0; v < 8; ++v) {
            int kb = half * 8 + c * 32 + ((v >= 4) ? 16 : 0) + 2 * (v & 3);
            A[c][2 * v]     = (_Float16)fmaf(ap[kb],     iv, epsl * hp[kb]);
            A[c][2 * v + 1] = (_Float16)fmaf(ap[kb + 1], iv, epsl * hp[kb + 1]);
        }
#pragma unroll
    for (int nt = 0; nt < 8; ++nt) {
        v8f acc = {};
        v16h B0 = loadB(sB + (((nt * 2 + 0) * 32 + L) * 8));
        v16h B1 = loadB(sB + (((nt * 2 + 1) * 32 + L) * 8));
        acc = __builtin_amdgcn_wmma_f32_16x16x32_f16(false, A[0], false, B0, (short)0, acc, false, false);
        acc = __builtin_amdgcn_wmma_f32_16x16x32_f16(false, A[1], false, B1, (short)0, acc, false, false);
        unsigned col = nt * 16u + row;
        float b = sbias[col];
        float s = 0.0f, q = 0.0f;
#pragma unroll
        for (int v = 0; v < 8; ++v) {
            unsigned nrow = tile * 16u + v + half * 8u;
            float yv = acc[v] + b;
            y[nrow * 128u + col] = yv;
            s += yv; q += yv * yv;
        }
        s += __shfl_xor(s, 16, 32);
        q += __shfl_xor(q, 16, 32);
        if (half == 0) { atomicAdd(&colsum[col], s); atomicAdd(&colsq[col], q); }
    }
}

__global__ void bnfin_k(const float* colsum, const float* colsq, const float* g, const float* be,
                        float* sc, float* sh) {
    int c = threadIdx.x; // 128
    float mean = colsum[c] * (1.0f / NN);
    float var  = colsq[c] * (1.0f / NN) - mean * mean;
    float s = g[c] * rsqrtf(var + BN_EPS);
    sc[c] = s;
    sh[c] = fmaf(-mean, s, be[c]);
}

// ---------------- node pass 2: z = relu(BN(y)) ; h' = relu(z@W_m2+b) ; write h and output slice ----------------
__global__ __launch_bounds__(256) void node2_k(const float* __restrict__ y, const float* __restrict__ sc,
                                               const float* __restrict__ sh, const unsigned* __restrict__ Bm2g,
                                               const float* __restrict__ bias,
                                               float* __restrict__ h, float* __restrict__ outp) {
    __shared__ unsigned sB[16 * 32 * 8];
    __shared__ float ssc[128], ssh[128], sb[64];
    int tid = threadIdx.x;
    for (int i = tid; i < 4096; i += 256) sB[i] = Bm2g[i];
    if (tid < 128) { ssc[tid] = sc[tid]; ssh[tid] = sh[tid]; }
    if (tid < 64) sb[tid] = bias[tid];
    __syncthreads();
    int wave = tid >> 5, L = tid & 31, row = L & 15, half = L >> 4;
    unsigned tile = blockIdx.x * 8u + wave;
    if (tile >= NN / 16) return;
    unsigned nA = tile * 16u + row;
    const float* yp = y + nA * 128u;
    v16h A[4];
#pragma unroll
    for (int c = 0; c < 4; ++c)
#pragma unroll
        for (int v = 0; v < 8; ++v) {
            int kb = half * 8 + c * 32 + ((v >= 4) ? 16 : 0) + 2 * (v & 3);
            A[c][2 * v]     = (_Float16)fmaxf(fmaf(yp[kb],     ssc[kb],     ssh[kb]),     0.0f);
            A[c][2 * v + 1] = (_Float16)fmaxf(fmaf(yp[kb + 1], ssc[kb + 1], ssh[kb + 1]), 0.0f);
        }
#pragma unroll
    for (int nt = 0; nt < 4; ++nt) {
        v8f acc = {};
#pragma unroll
        for (int kc = 0; kc < 4; ++kc) {
            v16h B = loadB(sB + (((nt * 4 + kc) * 32 + L) * 8));
            acc = __builtin_amdgcn_wmma_f32_16x16x32_f16(false, A[kc], false, B, (short)0, acc, false, false);
        }
        unsigned col = nt * 16u + row;
        float b = sb[col];
#pragma unroll
        for (int v = 0; v < 8; ++v) {
            unsigned nrow = tile * 16u + v + half * 8u;
            float o = fmaxf(acc[v] + b, 0.0f);
            h[nrow * 64u + col]    = o;
            outp[nrow * 256u + col] = o;
        }
    }
}

extern "C" void kernel_launch(void* const* d_in, const int* in_sizes, int n_in,
                              void* d_out, int out_size, void* d_ws, size_t ws_size,
                              hipStream_t stream) {
    (void)in_sizes; (void)n_in; (void)out_size; (void)ws_size;
    const float* x         = (const float*)d_in[0];
    const float* edge_attr = (const float*)d_in[1];
    const float* W_enc     = (const float*)d_in[2];
    const float* W_r1      = (const float*)d_in[4];
    const float* g_r       = (const float*)d_in[6];
    const float* be_r      = (const float*)d_in[7];
    const float* W_r2      = (const float*)d_in[8];
    const float* b_r2      = (const float*)d_in[9];
    const float* W_m1      = (const float*)d_in[10];
    const float* b_m1      = (const float*)d_in[11];
    const float* g_m       = (const float*)d_in[12];
    const float* be_m      = (const float*)d_in[13];
    const float* W_m2      = (const float*)d_in[14];
    const float* b_m2      = (const float*)d_in[15];
    const float* eps       = (const float*)d_in[16];
    const int*   ei        = (const int*)d_in[17];
    // b_enc, b_r1 cancel inside the edge BatchNorm (rank-1 collapse); batch unused (single graph).
    float* out = (float*)d_out;

    char* ws = (char*)d_ws;
    size_t off = 0;
    auto alloc = [&](size_t b) { size_t o = off; off = (off + b + 255) & ~(size_t)255; return o; };
    float*    h      = (float*)(ws + alloc((size_t)NN * 64 * 4));
    float*    agg    = (float*)(ws + alloc((size_t)NN * 64 * 4));
    float*    y      = (float*)(ws + alloc((size_t)NN * 128 * 4));
    float*    invd   = (float*)(ws + alloc((size_t)NN * 4));
    float*    stats  = (float*)(ws + alloc(2 * 4));
    float*    mav    = (float*)(ws + alloc(2 * 4));
    float*    svec   = (float*)(ws + alloc(64 * 4));
    float*    colsum = (float*)(ws + alloc(128 * 4));
    float*    colsq  = (float*)(ws + alloc(128 * 4));
    float*    bnsc   = (float*)(ws + alloc(128 * 4));
    float*    bnsh   = (float*)(ws + alloc(128 * 4));
    unsigned* Br2    = (unsigned*)(ws + alloc(8 * 32 * 8 * 4));
    unsigned* Bm1    = (unsigned*)(ws + alloc(16 * 32 * 8 * 4));
    unsigned* Bm2    = (unsigned*)(ws + alloc(16 * 32 * 8 * 4));

    // once per call: degrees, edge_attr stats, h = x
    zero_k<<<(NN + 255) / 256, 256, 0, stream>>>(invd, NN);
    zero_k<<<1, 256, 0, stream>>>(stats, 2);
    deg_k<<<(EE + 255) / 256, 256, 0, stream>>>(ei + EE, invd);
    degfin_k<<<(NN + 255) / 256, 256, 0, stream>>>(invd);
    eastats_k<<<256, 256, 0, stream>>>(edge_attr, stats);
    copy_k<<<(NN * 64 + 255) / 256, 256, 0, stream>>>(h, x, NN * 64);

    const int edgeBlocks = EE / 16 / 8;          // 6250, exact
    const int nodeBlocks = (NN / 16 + 7) / 8;    // 391 (guarded)
    for (int l = 0; l < LL; ++l) {
        zero_k<<<(NN * 64 + 255) / 256, 256, 0, stream>>>(agg, NN * 64);
        zero_k<<<1, 256, 0, stream>>>(colsum, 128);
        zero_k<<<1, 256, 0, stream>>>(colsq, 128);
        prep_k<<<1, 128, 0, stream>>>(l, W_enc, W_r1, g_r, W_r2, W_m1, W_m2,
                                      stats, mav, svec, Br2, Bm1, Bm2);
        edge_k<<<edgeBlocks, 256, 0, stream>>>(edge_attr, ei, mav, svec,
                                               be_r + l * 64, b_r2 + l * 64, Br2, h, agg);
        node1_k<<<nodeBlocks, 256, 0, stream>>>(h, agg, invd, eps + l, Bm1,
                                                b_m1 + l * 128, y, colsum, colsq);
        bnfin_k<<<1, 128, 0, stream>>>(colsum, colsq, g_m + l * 128, be_m + l * 128, bnsc, bnsh);
        node2_k<<<nodeBlocks, 256, 0, stream>>>(y, bnsc, bnsh, Bm2, b_m2 + l * 64,
                                                h, out + (size_t)l * 64);
    }
}
